// MGIGNN_59425167507634
// MI455X (gfx1250) — compile-verified, matrix operand order checked by reading, not run on hardware
//
#include <hip/hip_runtime.h>
#include <hip/hip_bf16.h>
#include <math.h>

// Problem constants (match reference)
#define N_NODES 10000
#define F_INPUT 512
#define C_DIM   128
#define NCLS    40
#define NE      160000
#define NLAY    4
#define NH      8
#define NG      16
#define KTOP    10
#define CG      8          // C/G
#define HD      16         // C/H
#define XSLOTS  (NLAY + 1) // x_all history slots

typedef __attribute__((ext_vector_type(2)))  float  v2f;
typedef __attribute__((ext_vector_type(8)))  float  v8f;
typedef __attribute__((ext_vector_type(8)))  __bf16 v8bf;
typedef __attribute__((ext_vector_type(16))) __bf16 v16bf;

// ---------------------------------------------------------------- degrees
__global__ void k_deg_init(float* __restrict__ deg) {
    int i = blockIdx.x * blockDim.x + threadIdx.x;
    if (i < N_NODES) deg[i] = 1.0f;              // self-loop contribution
}
__global__ void k_deg_acc(const int* __restrict__ col, float* __restrict__ deg) {
    int i = blockIdx.x * blockDim.x + threadIdx.x;
    if (i < NE) atomicAdd(&deg[col[i]], 1.0f);
}
__global__ void k_dinv(const float* __restrict__ deg, float* __restrict__ dinv) {
    int i = blockIdx.x * blockDim.x + threadIdx.x;
    if (i < N_NODES) dinv[i] = rsqrtf(fmaxf(deg[i], 1e-12f));
}
__global__ void k_zero(float* __restrict__ p, int n) {
    int i = blockIdx.x * blockDim.x + threadIdx.x;
    if (i < n) p[i] = 0.0f;
}

// ------------------------------------------------ h = relu(x@W1+b1), f32 WMMA
// One wave per 16x16 output tile; K = 512 in steps of 4 (v_wmma_f32_16x16x4_f32)
__global__ void __launch_bounds__(32)
k_gemm1(const float* __restrict__ x, const float* __restrict__ W1,
        const float* __restrict__ b1, float* __restrict__ x_all) {
    const int lane = threadIdx.x & 31;
    const int half = lane >> 4;
    const int l16  = lane & 15;
    const int rowBase = blockIdx.x * 16;     // 625 tiles
    const int colBase = blockIdx.y * 16;     // 8 tiles
    v8f acc = {0.f,0.f,0.f,0.f,0.f,0.f,0.f,0.f};
    const float* xrow = x + (size_t)(rowBase + l16) * F_INPUT + 2 * half;
    const float* wcol = W1 + colBase + l16;
    for (int kb = 0; kb < F_INPUT; kb += 4) {
        // A 16x4 f32: lane m=l16; element e -> K = kb + 2*half + e (contiguous pair)
        v2f a = *(const v2f*)(xrow + kb);
        // B 4x16: lane n=l16; element e -> K = kb + 2*half + e
        v2f b;
        b.x = wcol[(size_t)(kb + 2 * half) * C_DIM];
        b.y = wcol[(size_t)(kb + 2 * half + 1) * C_DIM];
        acc = __builtin_amdgcn_wmma_f32_16x16x4_f32(
            false, a, false, b, (short)0, acc, false, false);
    }
    const float bias = b1[colBase + l16];
#pragma unroll
    for (int r = 0; r < 8; ++r) {
        int m = rowBase + r + 8 * half;      // C/D layout: M = r + 8*half, N = l16
        x_all[(size_t)m * (XSLOTS * C_DIM) + colBase + l16] =
            fmaxf(acc[r] + bias, 0.0f);
    }
}

// ---------------------------------------- per-node grouped q/k/v precompute
// One thread per (node, history slot j, group g). q only for j == l.
__global__ void k_qkv(const float* __restrict__ x_all,
                      const float* __restrict__ Wq, const float* __restrict__ bq,
                      const float* __restrict__ Wk, const float* __restrict__ bk,
                      const float* __restrict__ Wv, const float* __restrict__ bv,
                      float* __restrict__ q_all, float* __restrict__ k_all,
                      float* __restrict__ v_all, int l) {
    const int L = l + 1;
    int tid = blockIdx.x * blockDim.x + threadIdx.x;
    if (tid >= N_NODES * L * NG) return;
    const int g = tid % NG;
    const int j = (tid / NG) % L;
    const int n = tid / (NG * L);
    float xi[CG];
    const float* xr = x_all + (size_t)n * (XSLOTS * C_DIM) + j * C_DIM + g * CG;
#pragma unroll
    for (int i = 0; i < CG; ++i) xi[i] = xr[i];
    const float* wk = Wk + g * CG * CG;
    const float* wv = Wv + g * CG * CG;
    float* kout = k_all + (size_t)n * (NLAY * C_DIM) + j * C_DIM + g * CG;
    float* vout = v_all + (size_t)n * (NLAY * C_DIM) + j * C_DIM + g * CG;
#pragma unroll
    for (int o = 0; o < CG; ++o) {
        float sk = bk[g * CG + o], sv = bv[g * CG + o];
#pragma unroll
        for (int i = 0; i < CG; ++i) {
            sk += xi[i] * wk[i * CG + o];
            sv += xi[i] * wv[i * CG + o];
        }
        kout[o] = sk;
        vout[o] = sv;
    }
    if (j == l) {
        const float* wq = Wq + g * CG * CG;
        float* qout = q_all + (size_t)n * C_DIM + g * CG;
#pragma unroll
        for (int o = 0; o < CG; ++o) {
            float sq = bq[g * CG + o];
#pragma unroll
            for (int i = 0; i < CG; ++i) sq += xi[i] * wq[i * CG + o];
            qout[o] = sq;
        }
    }
}

// -------------------------- per-(edge, head) attention + atomic scatter
__global__ void k_edge_attn(const int* __restrict__ ei,
                            const float* __restrict__ dinv,
                            const float* __restrict__ q_all,
                            const float* __restrict__ k_all,
                            const float* __restrict__ v_all,
                            float* __restrict__ agg, int l) {
    int tid = blockIdx.x * blockDim.x + threadIdx.x;
    if (tid >= (NE + N_NODES) * NH) return;
    const int h = tid % NH;
    const int e = tid / NH;
    const int row = (e < NE) ? ei[e]      : (e - NE);   // self loops appended
    const int col = (e < NE) ? ei[NE + e] : (e - NE);
    const float ew = dinv[row] * dinv[col];
    const int L = l + 1;
    float q[HD];
    const float* qp = q_all + (size_t)col * C_DIM + h * HD;
#pragma unroll
    for (int d = 0; d < HD; ++d) q[d] = qp[d];
    float sc[NLAY];
    float m = -3.0e38f;
#pragma unroll
    for (int j = 0; j < NLAY; ++j) {
        if (j < L) {
            const float* kp = k_all + (size_t)row * (NLAY * C_DIM) + j * C_DIM + h * HD;
            float s = 0.f;
#pragma unroll
            for (int d = 0; d < HD; ++d) s += q[d] * kp[d];
            s *= 0.25f;                       // 1/sqrt(HD=16)
            sc[j] = s;
            m = fmaxf(m, s);
        }
    }
    float denom = 0.f;
#pragma unroll
    for (int j = 0; j < NLAY; ++j)
        if (j < L) { sc[j] = expf(sc[j] - m); denom += sc[j]; }
    const float inv = 1.0f / denom;
    float out[HD];
#pragma unroll
    for (int d = 0; d < HD; ++d) out[d] = 0.f;
#pragma unroll
    for (int j = 0; j < NLAY; ++j) {
        if (j < L) {
            const float* vp = v_all + (size_t)row * (NLAY * C_DIM) + j * C_DIM + h * HD;
            const float a = sc[j] * inv;
#pragma unroll
            for (int d = 0; d < HD; ++d) out[d] += a * vp[d];
        }
    }
    float* ap = agg + (size_t)col * C_DIM + h * HD;
#pragma unroll
    for (int d = 0; d < HD; ++d) atomicAdd(&ap[d], ew * out[d]);
}

__global__ void k_relu_app(const float* __restrict__ agg, float* __restrict__ x_all, int l) {
    int i = blockIdx.x * blockDim.x + threadIdx.x;
    if (i >= N_NODES * C_DIM) return;
    const int n = i / C_DIM, c = i % C_DIM;
    x_all[(size_t)n * (XSLOTS * C_DIM) + (l + 1) * C_DIM + c] = fmaxf(agg[i], 0.0f);
}

// --------------------- emb output + L2-normalized bf16 rows for the sim GEMM
__global__ void __launch_bounds__(32)
k_embnorm(const float* __restrict__ x_all, float* __restrict__ emb_out,
          __bf16* __restrict__ xn) {
    const int n = blockIdx.x;
    const int lane = threadIdx.x;     // wave32
    const float* er = x_all + (size_t)n * (XSLOTS * C_DIM) + NLAY * C_DIM;
    float4 v = *(const float4*)(er + lane * 4);
    float ss = v.x * v.x + v.y * v.y + v.z * v.z + v.w * v.w;
    for (int off = 16; off > 0; off >>= 1) ss += __shfl_xor(ss, off, 32);
    const float inv = 1.0f / fmaxf(sqrtf(ss), 1e-8f);
    *(float4*)(emb_out + (size_t)n * C_DIM + lane * 4) = v;
    __bf16* xp = xn + (size_t)n * C_DIM + lane * 4;
    xp[0] = (__bf16)(v.x * inv); xp[1] = (__bf16)(v.y * inv);
    xp[2] = (__bf16)(v.z * inv); xp[3] = (__bf16)(v.w * inv);
}

// --------------------------------------------- classifier + log_softmax
__global__ void k_logits(const float* __restrict__ x_all, const float* __restrict__ W2,
                         const float* __restrict__ b2, float* __restrict__ logits) {
    int tid = blockIdx.x * blockDim.x + threadIdx.x;
    if (tid >= N_NODES * NCLS) return;
    const int n = tid / NCLS, c = tid % NCLS;
    const float* er = x_all + (size_t)n * (XSLOTS * C_DIM) + NLAY * C_DIM;
    float s = b2[c];
    for (int k = 0; k < C_DIM; ++k) s += er[k] * W2[k * NCLS + c];
    logits[tid] = s;
}
__global__ void k_logsoftmax(float* __restrict__ logits) {
    int n = blockIdx.x * blockDim.x + threadIdx.x;
    if (n >= N_NODES) return;
    float* r = logits + (size_t)n * NCLS;
    float m = -3.0e38f;
    for (int c = 0; c < NCLS; ++c) m = fmaxf(m, r[c]);
    float s = 0.f;
    for (int c = 0; c < NCLS; ++c) s += expf(r[c] - m);
    const float lse = logf(s) + m;
    for (int c = 0; c < NCLS; ++c) r[c] -= lse;
}

// ------ fused sim = xn@xn^T (bf16 WMMA) + streaming top-K + label-prop blend
// Block = 4 waves; 16 rows per block; column chunks of 64 (one 16x16 WMMA
// tile per wave, K=128 via 4x v_wmma_f32_16x16x32_bf16). Sim never hits HBM.
// A-tile fragments are loaded ONCE and held in registers across the whole
// column sweep; mask rows for the next chunk are prefetched (global_prefetch_b8)
// to hide HBM latency of the 400 MB mask stream (the roofline limiter here).
__global__ void __launch_bounds__(128)
k_sim_topk(const __bf16* __restrict__ xn, const float* __restrict__ mask,
           const int* __restrict__ y, const float* __restrict__ p_lc,
           float* __restrict__ out_final) {
    __shared__ float simbuf[16][68];
    const int tid  = threadIdx.x;
    const int w    = tid >> 5;
    const int lane = tid & 31;
    const int half = lane >> 4;
    const int l16  = lane & 15;
    const int rowBase = blockIdx.x * 16;

    float tv[KTOP]; int ti[KTOP];
#pragma unroll
    for (int k = 0; k < KTOP; ++k) { tv[k] = -3.0e38f; ti[k] = 0; }

    // Hoisted A fragments: invariant over the entire column sweep.
    // A 16x32 bf16 layout: elems 0-7 -> K = kb*32 + 8*half + e   (one b128 load)
    //                      elems 8-15 -> K = kb*32 + 16 + 8*half + (e-8)
    const v8bf* arow = (const v8bf*)(xn + (size_t)(rowBase + l16) * C_DIM);
    union V16 { v16bf v; v8bf h[2]; };
    V16 afrag[4];
#pragma unroll
    for (int kb = 0; kb < 4; ++kb) {
        afrag[kb].h[0] = arow[kb * 4 + half];
        afrag[kb].h[1] = arow[kb * 4 + 2 + half];
    }

    const float* mrow = mask + (size_t)(rowBase + (tid & 15)) * N_NODES;

    for (int c0 = 0; c0 < N_NODES; c0 += 64) {
        const int colBase = c0 + w * 16;
        __syncthreads();                       // protect simbuf from prior scan
        if (colBase < N_NODES) {
            const v8bf* brow = (const v8bf*)(xn + (size_t)(colBase + l16) * C_DIM);
            v8f acc = {0.f,0.f,0.f,0.f,0.f,0.f,0.f,0.f};
#pragma unroll
            for (int kb = 0; kb < 4; ++kb) {   // K base = kb*32
                V16 b;                         // B[k,n] = xn[colBase+n, k]
                b.h[0] = brow[kb * 4 + half];
                b.h[1] = brow[kb * 4 + 2 + half];
                acc = __builtin_amdgcn_wmma_f32_16x16x32_bf16(
                    false, afrag[kb].v, false, b.v, (short)0, acc, false, false);
            }
#pragma unroll
            for (int r = 0; r < 8; ++r)        // M = r + 8*half, N = l16
                simbuf[r + 8 * half][w * 16 + l16] = acc[r];
        }
        __syncthreads();
        if (tid < 16) {                        // thread t owns row rowBase+t
            // Prefetch next chunk's mask segment (2 cachelines cover 64 floats)
            if (c0 + 64 < N_NODES) {
                __builtin_prefetch(mrow + c0 + 64, 0, 1);
                __builtin_prefetch(mrow + c0 + 96, 0, 1);
            }
#pragma unroll 4
            for (int cc = 0; cc < 64; ++cc) {
                const int col = c0 + cc;
                if (col < N_NODES) {
                    const float val = simbuf[tid][cc] * mrow[col];
                    if (val > tv[KTOP - 1]) {  // bubble-insert (stable on ties)
                        float v = val; int id = col;
#pragma unroll
                        for (int k = 0; k < KTOP; ++k) {
                            if (v > tv[k]) {
                                float tf = tv[k]; tv[k] = v; v = tf;
                                int   tq = ti[k]; ti[k] = id; id = tq;
                            }
                        }
                    }
                }
            }
        }
    }
    if (tid < 16) {
        const int row = rowBase + tid;
        float ev[KTOP]; int cls[KTOP];
#pragma unroll
        for (int k = 0; k < KTOP; ++k) { ev[k] = expf(tv[k]); cls[k] = y[ti[k]]; }
        float fused[NCLS];
        float m = -3.0e38f;
#pragma unroll
        for (int c = 0; c < NCLS; ++c) {
            float f = 0.f;
#pragma unroll
            for (int k = 0; k < KTOP; ++k) f += (cls[k] == c) ? ev[k] : 0.f;
            fused[c] = f;
            m = fmaxf(m, f);
        }
        float s = 0.f;
#pragma unroll
        for (int c = 0; c < NCLS; ++c) s += expf(fused[c] - m);
        const float lse = logf(s) + m;
        const float* pl = p_lc + (size_t)row * NCLS;
        float* po = out_final + (size_t)row * NCLS;
#pragma unroll
        for (int c = 0; c < NCLS; ++c)
            po[c] = 0.5f * pl[c] + 0.5f * (fused[c] - lse);
    }
}

// ---------------------------------------------------------------- launcher
extern "C" void kernel_launch(void* const* d_in, const int* in_sizes, int n_in,
                              void* d_out, int out_size, void* d_ws, size_t ws_size,
                              hipStream_t stream) {
    const float* x    = (const float*)d_in[0];
    const float* mask = (const float*)d_in[1];
    const int*   ei   = (const int*)d_in[2];    // [2, E]
    const int*   y    = (const int*)d_in[3];
    const float* W1   = (const float*)d_in[4];
    const float* b1   = (const float*)d_in[5];
    const float* Wq   = (const float*)d_in[6];  // [NL, G, CG, CG]
    const float* bq   = (const float*)d_in[7];  // [NL, C]
    const float* Wk   = (const float*)d_in[8];
    const float* bk   = (const float*)d_in[9];
    const float* Wv   = (const float*)d_in[10];
    const float* bv   = (const float*)d_in[11];
    const float* W2   = (const float*)d_in[12];
    const float* b2   = (const float*)d_in[13];

    float* out_final = (float*)d_out;
    float* out_emb   = (float*)d_out + (size_t)N_NODES * NCLS;

    // workspace layout (floats); total ~81 MB
    float* ws    = (float*)d_ws;
    float* deg   = ws;                   // 10240
    float* dinv  = ws + 10240;           // 10240
    float* x_all = ws + 20480;           // N * 5 * C = 6,400,000
    float* q_all = ws + 6420480;         // N * C     = 1,280,000
    float* k_all = ws + 7700480;         // N * 4 * C = 5,120,000
    float* v_all = ws + 12820480;        // N * 4 * C = 5,120,000
    float* agg   = ws + 17940480;        // N * C     = 1,280,000
    float* plc   = ws + 19220480;        // N * NCLS  =   400,000
    __bf16* xn   = (__bf16*)(ws + 19620480); // N * C bf16

    const int B = 256;
    k_deg_init<<<(N_NODES + B - 1) / B, B, 0, stream>>>(deg);
    k_deg_acc<<<(NE + B - 1) / B, B, 0, stream>>>(ei + NE, deg);
    k_dinv<<<(N_NODES + B - 1) / B, B, 0, stream>>>(deg, dinv);

    k_gemm1<<<dim3(N_NODES / 16, C_DIM / 16), 32, 0, stream>>>(x, W1, b1, x_all);

    for (int l = 0; l < NLAY; ++l) {
        const int tq = N_NODES * (l + 1) * NG;
        k_qkv<<<(tq + B - 1) / B, B, 0, stream>>>(
            x_all, Wq + l * NG * CG * CG, bq + l * C_DIM,
            Wk + l * NG * CG * CG, bk + l * C_DIM,
            Wv + l * NG * CG * CG, bv + l * C_DIM,
            q_all, k_all, v_all, l);
        k_zero<<<(N_NODES * C_DIM + B - 1) / B, B, 0, stream>>>(agg, N_NODES * C_DIM);
        const int te = (NE + N_NODES) * NH;
        k_edge_attn<<<(te + B - 1) / B, B, 0, stream>>>(ei, dinv, q_all, k_all, v_all, agg, l);
        k_relu_app<<<(N_NODES * C_DIM + B - 1) / B, B, 0, stream>>>(agg, x_all, l);
    }

    k_embnorm<<<N_NODES, 32, 0, stream>>>(x_all, out_emb, xn);
    k_logits<<<(N_NODES * NCLS + B - 1) / B, B, 0, stream>>>(x_all, W2, b2, plc);
    k_logsoftmax<<<(N_NODES + B - 1) / B, B, 0, stream>>>(plc);
    k_sim_topk<<<N_NODES / 16, 128, 0, stream>>>(xn, mask, y, plc, out_final);
}
